// SpikingParrot_41489384080072
// MI455X (gfx1250) — compile-verified, matrix operand order checked by reading, not run on hardware
//
#include <hip/hip_runtime.h>
#include <hip/hip_bf16.h>
#include <math.h>

// ---------------------------------------------------------------------------
// SpikingParrot forward on MI455X (gfx1250, wave32, WMMA)
//   - dense GEMMs via V_WMMA_F32_16X16X32_F16 (f16 in, f32 accum)
//   - B-tiles staged in LDS with GLOBAL_LOAD_ASYNC_TO_LDS_B128 (ASYNCcnt),
//     double-buffered so WMMA overlaps the fill
//   - sequential SLSTM recurrence as a persistent single-WGP scan kernel
// ---------------------------------------------------------------------------

typedef __attribute__((ext_vector_type(16))) _Float16 v16h;
typedef __attribute__((ext_vector_type(8)))  _Float16 v8h;
typedef __attribute__((ext_vector_type(8)))  float    v8f;

#define DEV __device__ __forceinline__

static constexpr int Bc = 16;     // batch
static constexpr int Tc = 256;    // sequence length
static constexpr int Ec = 512;    // embedding dim
static constexpr int Hc = 512;    // encoder hidden
static constexpr int Vc = 32000;  // vocab
static constexpr int BT = Bc * Tc;

DEV float sigf(float x) { return 1.0f / (1.0f + __expf(-x)); }

// Load one 16x32 f16 fragment (A or B^T) from K-contiguous row-major storage.
// CDNA5 wave32 layout: lanes 0-15 -> row = lane, half[0..7]=K{0..7}, half[8..15]=K{16..23}
//                      lanes 16-31 -> row = lane-16, half[0..7]=K{8..15}, half[8..15]=K{24..31}
DEV v16h load_frag(const _Float16* base, int ld, int lane) {
  const int r    = lane & 15;
  const int koff = (lane & 16) ? 8 : 0;
  const _Float16* p = base + (size_t)r * ld + koff;
  union { v16h v; v8h h[2]; } u;
  u.h[0] = *(const v8h*)(p);        // 128-bit load
  u.h[1] = *(const v8h*)(p + 16);   // 128-bit load
  return u.v;
}

// Async 16-byte global -> LDS copy (per active lane). VDST carries the LDS
// byte address, VADDR the 64-bit global address (GV mode, no SADDR).
DEV void async_ld_b128(unsigned ldsAddr, const void* gaddr) {
  asm volatile("global_load_async_to_lds_b128 %0, %1, off"
               :: "v"(ldsAddr), "v"(gaddr) : "memory");
}
DEV void wait_async_le1() { asm volatile("s_wait_asynccnt 0x1" ::: "memory"); }
DEV void wait_async_0()   { asm volatile("s_wait_asynccnt 0x0" ::: "memory"); }

// ---------------------------------------------------------------------------
// GEMM: C[M,N] = A[M,K](f16) * W[N,K](f16)^T + bias
// block = 256 threads (8 waves) -> 128(M) x 64(N) tile.
// All 8 waves share the same 64 N-columns, so the 64x32 B-tile is staged once
// per block in LDS (async, double-buffered); A fragments stream from global.
// ---------------------------------------------------------------------------
__global__ __launch_bounds__(256)
void wmma_gemm(const _Float16* __restrict__ A, int lda,
               const _Float16* __restrict__ W, int ldw,
               const float* __restrict__ bias,
               float* __restrict__ C, int ldc,
               int M, int K) {
  __shared__ __align__(16) _Float16 Bs[2][64 * 32];   // 2 x 4 KB

  const int tid  = threadIdx.x;
  const int lane = tid & 31;
  const int wave = tid >> 5;
  const int m0   = blockIdx.y * 128 + wave * 16;
  const bool active = (m0 < M);                 // wave-uniform
  const int nb = blockIdx.x * 64;

  // cooperative fill map: thread -> one 16B chunk of the 64x32 B-tile
  const int frow = tid >> 2;                    // 0..63 (N within tile)
  const int fk   = (tid & 3) * 8;               // 0,8,16,24 (K within chunk)
  const _Float16* wRow = W + (size_t)(nb + frow) * ldw + fk;

  v8f acc[4] = {v8f{}, v8f{}, v8f{}, v8f{}};
  const _Float16* aRow = active ? (A + (size_t)m0 * lda) : A;

  // prologue: fill buffer 0 with k0 = 0
  async_ld_b128((unsigned)(uintptr_t)&Bs[0][frow * 32 + fk], wRow);

  for (int k0 = 0; k0 < K; k0 += 32) {
    const int cur = (k0 >> 5) & 1;
    const bool more = (k0 + 32 < K);
    if (more) {  // fill the other buffer with the next K-slice
      async_ld_b128((unsigned)(uintptr_t)&Bs[cur ^ 1][frow * 32 + fk],
                    wRow + (k0 + 32));
      wait_async_le1();          // my older fill done; newest may be in flight
    } else {
      wait_async_0();
    }
    __syncthreads();             // every wave finished filling its chunks

    if (active) {
      __builtin_prefetch(aRow + k0 + 32, 0, 3);         // global_prefetch_b8
      v16h a = load_frag(aRow + k0, lda, lane);
#pragma unroll
      for (int j = 0; j < 4; ++j) {
        v16h b = load_frag(&Bs[cur][16 * j * 32], 32, lane);   // ds_load_b128
        acc[j] = __builtin_amdgcn_wmma_f32_16x16x32_f16(
            false, a, false, b, (short)0, acc[j], false, false);
      }
    }
    __syncthreads();             // tile fully consumed before it is refilled
  }

  if (active) {
    // C/D layout: VGPR r -> row r (lanes 0-15) / r+8 (lanes 16-31); lane%16 -> col
    const int col = lane & 15;
    const int rb  = (lane & 16) ? 8 : 0;
#pragma unroll
    for (int j = 0; j < 4; ++j) {
      const int n = nb + 16 * j + col;
      const float bv = bias ? bias[n] : 0.0f;
#pragma unroll
      for (int r = 0; r < 8; ++r) {
        C[(size_t)(m0 + rb + r) * ldc + n] = acc[j][r] + bv;
      }
    }
  }
}

// ---------------------------------------------------------------------------
// Persistent SLSTM scan: one workgroup of 1024 threads (32 waves) runs all
// T steps for one layer.  Per step:
//   phase A: gates[16,4H] = mem[16,H](LDS,f16) @ Whh[4H,H]^T  via WMMA -> LDS
//   phase B: per-thread LSTM update; syn(c)/mem(h) live in registers,
//            new mem written back to LDS (f16) for the next step's WMMA.
// Gin holds x@Wih^T + b_ih precomputed for all timesteps.
// ---------------------------------------------------------------------------
__global__ __launch_bounds__(1024)
void slstm_scan(const float* __restrict__ Gin,      // [B*T, 4H]
                const _Float16* __restrict__ Whh,   // [4H, H] f16
                const float* __restrict__ bhh,      // [4H]
                const float* __restrict__ thrP,     // scalar
                const float* __restrict__ syn0,     // [B*H] or nullptr
                const float* __restrict__ mem0,     // [B*H] or nullptr
                _Float16* __restrict__ spks,        // [B*T, H]
                float* __restrict__ synOut,         // [B*H]
                float* __restrict__ memOut,         // [B*H]
                int H) {
  extern __shared__ char smem[];
  const int N4 = 4 * H;
  float*    gS = (float*)smem;                                   // 16 x 4H gates
  _Float16* mS = (_Float16*)(smem + (size_t)16 * N4 * sizeof(float)); // 16 x H mem

  const int tid  = threadIdx.x;
  const int lane = tid & 31;
  const int wave = tid >> 5;
  const float thr = thrP[0];

  const int U = (16 * H) / 1024;  // units per thread: 8 (H=512) or 16 (H=1024)
  float c[16], m[16];
  for (int j = 0; j < U; ++j) {
    const int u = j * 1024 + tid;            // coalesced ownership map
    c[j] = syn0 ? syn0[u] : 0.0f;
    m[j] = mem0 ? mem0[u] : 0.0f;
    mS[u] = (_Float16)m[j];
  }
  __syncthreads();

  const int tilesPerWave = N4 / 16 / 32;     // 4 (H=512) or 8 (H=1024)

  for (int t = 0; t < Tc; ++t) {
    // ---- phase A: recurrent GEMM via WMMA (all waves, full EXEC) ----
    for (int jt = 0; jt < tilesPerWave; ++jt) {
      const int n0 = (jt * 32 + wave) * 16;
      v8f acc = {};
      for (int k0 = 0; k0 < H; k0 += 32) {
        v16h a = load_frag(mS + k0, H, lane);                       // LDS reads
        v16h b = load_frag(Whh + (size_t)n0 * H + k0, H, lane);     // L2-resident
        acc = __builtin_amdgcn_wmma_f32_16x16x32_f16(
            false, a, false, b, (short)0, acc, false, false);
      }
      const int col = lane & 15;
      const int rb  = (lane & 16) ? 8 : 0;
#pragma unroll
      for (int r = 0; r < 8; ++r)
        gS[(size_t)(rb + r) * N4 + n0 + col] = acc[r];
    }
    __syncthreads();

    // ---- phase B: elementwise LSTM / spike / subtract-reset ----
    for (int j = 0; j < U; ++j) {
      const int u = j * 1024 + tid;
      const int b = u / H, h = u - b * H;
      const size_t bt = (size_t)b * Tc + t;
      const float* gi = Gin + bt * N4;
      const float* gl = gS + (size_t)b * N4;
      const float ig = gl[h]         + gi[h]         + bhh[h];
      const float fg = gl[H + h]     + gi[H + h]     + bhh[H + h];
      const float gg = gl[2 * H + h] + gi[2 * H + h] + bhh[2 * H + h];
      const float og = gl[3 * H + h] + gi[3 * H + h] + bhh[3 * H + h];
      const float reset = (m[j] - thr) > 0.0f ? 1.0f : 0.0f;  // from pre-step mem
      c[j] = sigf(fg) * c[j] + sigf(ig) * tanhf(gg);
      const float hv = sigf(og) * tanhf(c[j]);
      m[j] = hv - reset * thr;
      const float spk = (m[j] - thr) > 0.0f ? 1.0f : 0.0f;    // Heaviside fwd
      spks[bt * H + h] = (_Float16)spk;
      mS[u] = (_Float16)m[j];
    }
    __syncthreads();
  }

  for (int j = 0; j < U; ++j) {
    const int u = j * 1024 + tid;
    synOut[u] = c[j];
    memOut[u] = m[j];
  }
}

// ---------------------------------------------------------------------------
// Elementwise kernels
// ---------------------------------------------------------------------------
__global__ void k_cvt(const float* __restrict__ a, _Float16* __restrict__ o, size_t n) {
  for (size_t i = blockIdx.x * (size_t)blockDim.x + threadIdx.x; i < n;
       i += (size_t)gridDim.x * blockDim.x)
    o[i] = (_Float16)a[i];
}

// out[bt, e] = dyt(emb[ids[b, t or T-1-t], e])   (f16 output)
__global__ void k_embed_dyt(const int* __restrict__ ids, const float* __restrict__ emb,
                            const float* __restrict__ alphaP, const float* __restrict__ gamma,
                            const float* __restrict__ beta, _Float16* __restrict__ out,
                            int flip) {
  const float alpha = alphaP[0];
  const size_t n = (size_t)BT * Ec;
  for (size_t i = blockIdx.x * (size_t)blockDim.x + threadIdx.x; i < n;
       i += (size_t)gridDim.x * blockDim.x) {
    const int bt = (int)(i / Ec), e = (int)(i % Ec);
    const int b = bt / Tc, t = bt % Tc;
    const int tok = ids[b * Tc + (flip ? (Tc - 1 - t) : t)];
    const float v = emb[(size_t)tok * Ec + e];
    out[i] = (_Float16)(gamma[e] * tanhf(alpha * v) + beta[e]);
  }
}

// out = f16(dyt(tmp (+ x if residual)))
__global__ void k_res_dyt(const float* __restrict__ tmp, const _Float16* __restrict__ x,
                          const float* __restrict__ alphaP, const float* __restrict__ gamma,
                          const float* __restrict__ beta, _Float16* __restrict__ out,
                          size_t n, int C, int residual) {
  const float alpha = alphaP[0];
  for (size_t i = blockIdx.x * (size_t)blockDim.x + threadIdx.x; i < n;
       i += (size_t)gridDim.x * blockDim.x) {
    const int c = (int)(i % C);
    float v = tmp[i];
    if (residual) v += (float)x[i];
    out[i] = (_Float16)(gamma[c] * tanhf(alpha * v) + beta[c]);
  }
}

// out[b, 0..half-1]=a[b,:], out[b, half..2half-1]=bb[b,:]  (f32 -> f16)
__global__ void k_concat_f16(const float* __restrict__ a, const float* __restrict__ bb,
                             _Float16* __restrict__ o, int half) {
  const size_t n = (size_t)Bc * 2 * half;
  for (size_t i = blockIdx.x * (size_t)blockDim.x + threadIdx.x; i < n;
       i += (size_t)gridDim.x * blockDim.x) {
    const int b = (int)(i / (2 * half)), c = (int)(i % (2 * half));
    o[i] = (_Float16)(c < half ? a[(size_t)b * half + c] : bb[(size_t)b * half + c - half]);
  }
}

// ---------------------------------------------------------------------------
// Host orchestration
// ---------------------------------------------------------------------------
extern "C" void kernel_launch(void* const* d_in, const int* in_sizes, int n_in,
                              void* d_out, int out_size, void* d_ws, size_t ws_size,
                              hipStream_t stream) {
  (void)in_sizes; (void)n_in; (void)out_size; (void)ws_size;

  // ---- flat input index map (setup_inputs dict order, depth-first) ----
  // 0:src 1:tgt 2:src_emb 3:tgt_emb 4-6:src_dyt(a,g,b) 7-9:tgt_dyt(a,g,b)
  // 10..19 enc_fwd[0], 20..29 enc_fwd[1], 30..39 enc_bwd[0], 40..49 enc_bwd[1],
  // 50..59 dec[0], 60..69 dec[1]   (per layer: w_ih,w_hh,b_ih,b_hh,thr,fc_w,fc_b,dyt a,g,b)
  // 70:syn_fc.w 71:syn_fc.b 72:mem_fc.w 73:mem_fc.b 74:out_fc.w 75:out_fc.b
  const int*   src     = (const int*)d_in[0];
  const int*   tgt     = (const int*)d_in[1];
  const float* src_emb = (const float*)d_in[2];
  const float* tgt_emb = (const float*)d_in[3];

  auto LP = [&](int base, const float** p) {
    for (int i = 0; i < 10; ++i) p[i] = (const float*)d_in[base + i];
  };
  const float* synfc_w = (const float*)d_in[70];
  const float* synfc_b = (const float*)d_in[71];
  const float* memfc_w = (const float*)d_in[72];
  const float* memfc_b = (const float*)d_in[73];
  const float* outfc_w = (const float*)d_in[74];
  const float* outfc_b = (const float*)d_in[75];

  // ---- workspace bump allocator ----
  char* cur = (char*)d_ws;
  auto alloc = [&](size_t bytes) -> void* {
    void* p = cur; cur += (bytes + 255) & ~(size_t)255; return p;
  };
  float*    Gin    = (float*)alloc((size_t)BT * 4096 * sizeof(float));     // gate pre-acts
  float*    tmpf   = (float*)alloc((size_t)BT * 1024 * sizeof(float));     // fc output
  _Float16* actA   = (_Float16*)alloc((size_t)BT * 1024 * 2);
  _Float16* actB   = (_Float16*)alloc((size_t)BT * 1024 * 2);
  _Float16* spk16  = (_Float16*)alloc((size_t)BT * 1024 * 2);
  _Float16* wih16  = (_Float16*)alloc((size_t)4096 * 1024 * 2);
  _Float16* whh16  = (_Float16*)alloc((size_t)4096 * 1024 * 2);
  _Float16* fcw16  = (_Float16*)alloc((size_t)1024 * 1024 * 2);
  _Float16* outw16 = (_Float16*)alloc((size_t)Vc * 1024 * 2);
  _Float16* sfw16  = (_Float16*)alloc((size_t)1024 * 1024 * 2);
  _Float16* mfw16  = (_Float16*)alloc((size_t)1024 * 1024 * 2);
  float* synF = (float*)alloc((size_t)Bc * Hc * 4);
  float* memF = (float*)alloc((size_t)Bc * Hc * 4);
  float* synB = (float*)alloc((size_t)Bc * Hc * 4);
  float* memB = (float*)alloc((size_t)Bc * Hc * 4);
  float* synI = (float*)alloc((size_t)Bc * 1024 * 4);   // decoder initial states
  float* memI = (float*)alloc((size_t)Bc * 1024 * 4);
  float* sScr = (float*)alloc((size_t)Bc * 1024 * 4);   // discarded states
  float* mScr = (float*)alloc((size_t)Bc * 1024 * 4);
  _Float16* catS = (_Float16*)alloc((size_t)Bc * 1024 * 2);
  _Float16* catM = (_Float16*)alloc((size_t)Bc * 1024 * 2);

  // ---- launch helpers ----
  auto cvt = [&](const float* a, _Float16* o, size_t n) {
    int blocks = (int)((n + 2047) / 2048); if (blocks > 4096) blocks = 4096;
    k_cvt<<<blocks, 256, 0, stream>>>(a, o, n);
  };
  auto gemm = [&](const _Float16* A, int lda, const _Float16* W, int ldw,
                  const float* bias, float* C, int N, int M, int K) {
    dim3 g(N / 64, (M + 127) / 128);
    wmma_gemm<<<g, 256, 0, stream>>>(A, lda, W, ldw, bias, C, N, M, K);
  };
  auto scan = [&](const float* gin, const _Float16* whh, const float* bhh,
                  const float* thr, const float* s0, const float* m0,
                  _Float16* spk, float* so, float* mo, int hid) {
    size_t sm = (size_t)16 * 4 * hid * sizeof(float) + (size_t)16 * hid * 2;
    slstm_scan<<<1, 1024, sm, stream>>>(gin, whh, bhh, thr, s0, m0, spk, so, mo, hid);
  };
  auto ew = [&](size_t n) { int b = (int)((n + 2047) / 2048); return b > 4096 ? 4096 : b; };

  auto run_layer = [&](const float* const* P, const _Float16* x16, _Float16* out16,
                       int in_dim, int hid, int residual,
                       const float* s0, const float* m0, float* so, float* mo) {
    cvt(P[0], wih16, (size_t)4 * hid * in_dim);
    cvt(P[1], whh16, (size_t)4 * hid * hid);
    cvt(P[5], fcw16, (size_t)hid * hid);
    // Gin = x @ w_ih^T + b_ih   (all T at once)
    gemm(x16, in_dim, wih16, in_dim, P[2], Gin, 4 * hid, BT, in_dim);
    // sequential scan (adds mem@w_hh^T + b_hh, LSTM gates, spikes, reset)
    scan(Gin, whh16, P[3], P[4], s0, m0, spk16, so, mo, hid);
    // out = dyt(spks @ fc_w^T + fc_b (+ x))
    gemm(spk16, hid, fcw16, hid, P[6], tmpf, hid, BT, hid);
    k_res_dyt<<<ew((size_t)BT * hid), 256, 0, stream>>>(
        tmpf, x16, P[7], P[8], P[9], out16, (size_t)BT * hid, hid, residual);
  };

  const float* P[10];

  // ===== encoder forward =====
  k_embed_dyt<<<ew((size_t)BT * Ec), 256, 0, stream>>>(
      src, src_emb, (const float*)d_in[4], (const float*)d_in[5], (const float*)d_in[6],
      actA, /*flip=*/0);
  LP(10, P); run_layer(P, actA, actB, Ec, Hc, 0, nullptr, nullptr, sScr, mScr);
  LP(20, P); run_layer(P, actB, actA, Hc, Hc, 1, nullptr, nullptr, synF, memF);

  // ===== encoder backward (time-flipped input) =====
  k_embed_dyt<<<ew((size_t)BT * Ec), 256, 0, stream>>>(
      src, src_emb, (const float*)d_in[4], (const float*)d_in[5], (const float*)d_in[6],
      actA, /*flip=*/1);
  LP(30, P); run_layer(P, actA, actB, Ec, Hc, 0, nullptr, nullptr, sScr, mScr);
  LP(40, P); run_layer(P, actB, actA, Hc, Hc, 1, nullptr, nullptr, synB, memB);

  // ===== bridge: syn/mem = concat(f,b) @ {syn,mem}_fc + b =====
  k_concat_f16<<<ew((size_t)Bc * 1024), 256, 0, stream>>>(synF, synB, catS, Hc);
  k_concat_f16<<<ew((size_t)Bc * 1024), 256, 0, stream>>>(memF, memB, catM, Hc);
  cvt(synfc_w, sfw16, (size_t)1024 * 1024);
  cvt(memfc_w, mfw16, (size_t)1024 * 1024);
  gemm(catS, 1024, sfw16, 1024, synfc_b, synI, 1024, Bc, 1024);
  gemm(catM, 1024, mfw16, 1024, memfc_b, memI, 1024, Bc, 1024);

  // ===== decoder =====
  k_embed_dyt<<<ew((size_t)BT * Ec), 256, 0, stream>>>(
      tgt, tgt_emb, (const float*)d_in[7], (const float*)d_in[8], (const float*)d_in[9],
      actA, /*flip=*/0);
  LP(50, P); run_layer(P, actA, actB, Ec,   1024, 0, synI, memI, sScr, mScr);
  LP(60, P); run_layer(P, actB, actA, 1024, 1024, 1, nullptr, nullptr, sScr, mScr);

  // ===== output projection: d_out[BT, V] = y @ out_fc.w^T + b =====
  cvt(outfc_w, outw16, (size_t)Vc * 1024);
  gemm(actA, 1024, outw16, 1024, outfc_b, (float*)d_out, Vc, BT, 1024);
}
